// similarity_uncertainty_71193377898740
// MI455X (gfx1250) — matrix operand order repaired and checked by reading, not compiled, not run on hardware
//
#include <hip/hip_runtime.h>
#include <hip/hip_bf16.h>
#include <math.h>

typedef __attribute__((ext_vector_type(16))) __bf16 v16bf;
typedef __attribute__((ext_vector_type(8)))  __bf16 v8bf;
typedef __attribute__((ext_vector_type(8)))  float  v8f;

#define GRD_K   131072                 // 32*64*64 per sample
#define SATH    160                    // padded sat height (64 data + pads + tile slack)
#define SATW    160                    // padded sat width
#define GPW     128                    // padded grd row width (x offset +16)
#define HO      65                     // output plane side
#define NT      5                      // 16-wide tiles per side
#define SATPLANE (32 * SATH * SATW)    // elems per sat map
#define GPLANE   (32 * 64 * GPW)       // elems per padded grd template

// ---------------- K1: per-sample 1/max(||x||,eps) --------------------------
__global__ void k_norms(const float* __restrict__ grd, const float* __restrict__ sat,
                        float* __restrict__ rn) {
  int s = blockIdx.x;  // 0..3 -> grd, 4..35 -> sat
  const float* src = (s < 4) ? (grd + (size_t)s * GRD_K)
                             : (sat + (size_t)(s - 4) * GRD_K);
  float ss = 0.f;
  for (int i = threadIdx.x; i < GRD_K; i += blockDim.x) { float v = src[i]; ss += v * v; }
  __shared__ float red[256];
  red[threadIdx.x] = ss;
  __syncthreads();
  for (int off = 128; off > 0; off >>= 1) {
    if ((int)threadIdx.x < off) red[threadIdx.x] += red[threadIdx.x + off];
    __syncthreads();
  }
  if (threadIdx.x == 0) rn[s] = 1.0f / fmaxf(sqrtf(red[0]), 1e-12f);
}

// ---------------- K2a: normalized grd -> bf16, zero-padded rows (width 128)
// grd_pad[n][c][y][xi] = grd[n][c][y][xi-16] for xi in [16,80), else 0
__global__ void k_grd_pad(const float* __restrict__ grd, const float* __restrict__ rn,
                          __bf16* __restrict__ out) {
  int i = blockIdx.x * blockDim.x + threadIdx.x;
  if (i >= 4 * GPLANE) return;
  int xi = i & 127; int t = i >> 7;
  int y  = t & 63;  t >>= 6;
  int c  = t & 31;  int n = t >> 5;
  float v = 0.f;
  if (xi >= 16 && xi < 80)
    v = grd[(((size_t)n * 32 + c) * 64 + y) * 64 + (xi - 16)] * rn[n];
  out[i] = (__bf16)v;
}

// ---------------- K2b: normalized, zero-padded sat -> bf16 (160x160) -------
__global__ void k_sat_bf(const float* __restrict__ sat, const float* __restrict__ rn,
                         __bf16* __restrict__ out) {
  long long i = (long long)blockIdx.x * blockDim.x + threadIdx.x;
  const long long total = 32LL * SATPLANE;
  if (i >= total) return;
  int xx = (int)(i % SATW); long long t = i / SATW;
  int yy = (int)(t % SATH); t /= SATH;
  int c  = (int)(t % 32);
  int m  = (int)(t / 32);
  float v = 0.f;
  if (yy >= 32 && yy < 96 && xx >= 32 && xx < 96) {
    v = sat[(((size_t)m * 32 + c) * 64 + (yy - 32)) * 64 + (xx - 32)] * rn[4 + m];
  }
  out[i] = (__bf16)v;
}

// ---------------- K2c: s2[m][yy][xx] (128x128) = sum_c sat_norm^2 ----------
__global__ void k_s2(const float* __restrict__ sat, const float* __restrict__ rn,
                     float* __restrict__ s2) {
  int i = blockIdx.x * blockDim.x + threadIdx.x;
  const int total = 32 * 128 * 128;
  if (i >= total) return;
  int xx = i & 127; int t = i >> 7;
  int yy = t & 127; int m = t >> 7;
  float s = 0.f;
  if (yy >= 32 && yy < 96 && xx >= 32 && xx < 96) {
    float r = rn[4 + m]; r = r * r;
    const float* p = sat + ((size_t)m * 32 * 64 + (yy - 32)) * 64 + (xx - 32);
    for (int c = 0; c < 32; ++c) { float v = p[(size_t)c * 4096]; s += v * v; }
    s *= r;
  }
  s2[i] = s;
}

// ---------------- K3a: horizontal 64-window sums ---------------------------
__global__ void k_rowsum(const float* __restrict__ s2, float* __restrict__ rowsum) {
  int i = blockIdx.x * blockDim.x + threadIdx.x;
  const int total = 32 * 128 * HO;
  if (i >= total) return;
  int dx = i % HO; int t = i / HO;
  int yy = t & 127; int m = t >> 7;
  const float* p = s2 + (((size_t)m << 7) + yy) * 128 + dx;
  float s = 0.f;
  for (int x = 0; x < 64; ++x) s += p[x];
  rowsum[i] = s;
}

// ---------------- K3b: denominator field -----------------------------------
__global__ void k_denom(const float* __restrict__ rowsum, const int* __restrict__ unc,
                        float* __restrict__ denom) {
  int i = blockIdx.x * blockDim.x + threadIdx.x;
  const int total = 32 * HO * HO;
  if (i >= total) return;
  int dx = i % HO; int t = i / HO;
  int dy = t % HO; int m = t / HO;
  const float* p = rowsum + (((size_t)m << 7) + dy) * HO + dx;
  float s = 0.f;
  for (int y = 0; y < 64; ++y) s += p[(size_t)y * HO];
  float u = (float)unc[0];
  denom[i] = fmaxf(sqrtf(s) * u, 1e-12f);
}

// ---------------- K4: WMMA correlation, 16dy x 16dx tile per (m,n) ---------
// A rows = 16 vertical shifts (sat, aligned loads); B cols = 16 horizontal
// shifts (zero-padded grd, unaligned 32B slices). Double-buffered.
__global__ void __launch_bounds__(32)
k_corr(const __bf16* __restrict__ grd_pad, const __bf16* __restrict__ sat_bf,
       const float* __restrict__ denom, float* __restrict__ pvals,
       int* __restrict__ pidx) {
  const int tp  = blockIdx.x;                 // 0..24
  const int dyt = tp / NT, dxt = tp % NT;
  const int n = blockIdx.y, m = blockIdx.z;
  const int dy0 = dyt * 16, dx0 = dxt * 16;
  const int L = threadIdx.x, lo = L & 15, hi = L >> 4;

  v8f acc = {};
  // A: lane row dyr = lo; fixed lane offset (dy0+lo)*SATW + dx0 + hi*16 (8-elem aligned)
  const __bf16* ap = sat_bf + (size_t)m * SATPLANE
                   + (size_t)(dy0 + lo) * SATW + dx0 + hi * 16;
  // B: lane col dxc = lo; element index = xc*32 + hi*16 + j - lo + 16
  const __bf16* bp = grd_pad + (size_t)n * GPLANE + (hi * 16 - lo + 16);

  auto load_a = [](const __bf16* g) -> v16bf {
    v8bf s0 = *(const v8bf*)g;
    v8bf s1 = *(const v8bf*)(g + 8);
    v16bf r;
#pragma unroll
    for (int i = 0; i < 8; ++i) { r[i] = s0[i]; r[i + 8] = s1[i]; }
    return r;
  };
  auto load_b = [](const __bf16* s) -> v16bf {
    v16bf r; __builtin_memcpy(&r, s, 32); return r;
  };
  auto a_at = [&](int cy) -> const __bf16* {
    int c = cy >> 6, y = cy & 63;
    return ap + ((size_t)c * SATH + y) * SATW;
  };
  auto b_at = [&](int cy) -> const __bf16* {
    return bp + (size_t)cy * GPW;
  };

  // prologue: 3 x-chunks (x' window of 96) for cy = 0
  const __bf16* ac = a_at(0); const __bf16* bc = b_at(0);
  v16bf a0 = load_a(ac), a1 = load_a(ac + 32), a2 = load_a(ac + 64);
  v16bf b0 = load_b(bc), b1 = load_b(bc + 32), b2 = load_b(bc + 64);

#pragma unroll 2
  for (int cy = 0; cy < 2048; ++cy) {
    // next (c,y) row's fragments first (cy==2047 over-reads into ws; unused)
    const __bf16* an = a_at(cy + 1);
    const __bf16* bn = b_at(cy + 1);
    __builtin_prefetch(an + SATW, 0, 3);
    v16bf na0 = load_a(an), na1 = load_a(an + 32), na2 = load_a(an + 64);
    v16bf nb0 = load_b(bn), nb1 = load_b(bn + 32), nb2 = load_b(bn + 64);
    acc = __builtin_amdgcn_wmma_f32_16x16x32_bf16(
        false, a0, false, b0, (short)0, acc, false, false);
    acc = __builtin_amdgcn_wmma_f32_16x16x32_bf16(
        false, a1, false, b1, (short)0, acc, false, false);
    acc = __builtin_amdgcn_wmma_f32_16x16x32_bf16(
        false, a2, false, b2, (short)0, acc, false, false);
    a0 = na0; a1 = na1; a2 = na2;
    b0 = nb0; b1 = nb1; b2 = nb2;
  }

  // epilogue: lane L holds dx = dx0+lo, rows dy = dy0 + hi*8 + d (d = 0..7)
  float bestv = -INFINITY; int besti = 0x7fffffff;
  const int dxv = dx0 + lo;
  if (dxv < HO) {
#pragma unroll
    for (int d = 0; d < 8; ++d) {
      int dyv = dy0 + hi * 8 + d;
      if (dyv < HO) {
        float den = denom[((size_t)m * HO + dyv) * HO + dxv];
        float v = acc[d] / den;
        if (v > bestv) { bestv = v; besti = dyv * HO + dxv; }  // d asc => idx asc
      }
    }
  }
  __shared__ float svv[32];
  __shared__ int   svi[32];
  svv[L] = bestv; svi[L] = besti;
  __syncthreads();
  if (L == 0) {
    float bv = -INFINITY; int bi = 0x7fffffff;
    for (int j = 0; j < 32; ++j) {
      float v = svv[j]; int id = svi[j];
      if (v > bv || (v == bv && id < bi)) { bv = v; bi = id; }
    }
    const int slot = (((m * 4 + n) * NT) + dyt) * NT + dxt;
    pvals[slot] = bv; pidx[slot] = bi;
  }
}

// ---------------- K5: global argmax + output -------------------------------
__global__ void k_final(const float* __restrict__ pvals, const int* __restrict__ pidx,
                        float* __restrict__ out) {
  int t = threadIdx.x;              // 0..127
  int m = t >> 2, n = t & 3;
  float best = -INFINITY; int bidx = 0x7fffffff;
  for (int s = 0; s < NT * NT; ++s) {
    int off = (m * 4 + n) * NT * NT + s;
    float v = pvals[off]; int id = pidx[off];
    if (v > best || (v == best && id < bidx)) { best = v; bidx = id; }
  }
  out[m * 4 + n] = best;
  float row = (float)(bidx / HO), col = (float)(bidx % HO);
  const float half  = 32.5f;                 // 65/2
  const float scale = 0.2f * 512.f / 128.f;  // meter_per_pixel*sidelen/W_i = 0.8
  out[128 + (m * 4 + n) * 2 + 0] = -(row - half) * scale;
  out[128 + (m * 4 + n) * 2 + 1] =  (col - half) * scale;
}

// ---------------------------------------------------------------------------
extern "C" void kernel_launch(void* const* d_in, const int* in_sizes, int n_in,
                              void* d_out, int out_size, void* d_ws, size_t ws_size,
                              hipStream_t stream) {
  const float* grd = (const float*)d_in[0];  // [4,32,64,64]
  const float* sat = (const float*)d_in[1];  // [32,32,64,64]
  const int*   unc = (const int*)d_in[2];    // scalar
  float* out = (float*)d_out;                // 128 sims + 256 shifts

  char* ws = (char*)d_ws;
  size_t off = 0;
  auto alloc = [&](size_t bytes) -> char* {
    char* p = ws + off;
    off += (bytes + 255) & ~(size_t)255;
    return p;
  };
  float*  rn      = (float*) alloc(36 * sizeof(float));
  __bf16* grd_pad = (__bf16*)alloc((size_t)4 * GPLANE * 2);
  __bf16* sat_bf  = (__bf16*)alloc((size_t)32 * SATPLANE * 2);
  float*  s2      = (float*) alloc((size_t)32 * 128 * 128 * 4);
  float*  rowsum  = (float*) alloc((size_t)32 * 128 * HO * 4);
  float*  denom   = (float*) alloc((size_t)32 * HO * HO * 4);
  float*  pvals   = (float*) alloc((size_t)32 * 4 * NT * NT * 4);
  int*    pidx    = (int*)   alloc((size_t)32 * 4 * NT * NT * 4);
  (void)   alloc(65536);   // tail pad for the pipelined over-read

  k_norms<<<36, 256, 0, stream>>>(grd, sat, rn);

  {
    int tot = 4 * GPLANE;
    k_grd_pad<<<(tot + 255) / 256, 256, 0, stream>>>(grd, rn, grd_pad);
  }
  {
    long long tot = 32LL * SATPLANE;
    k_sat_bf<<<(int)((tot + 255) / 256), 256, 0, stream>>>(sat, rn, sat_bf);
  }
  {
    int tot = 32 * 128 * 128;
    k_s2<<<(tot + 255) / 256, 256, 0, stream>>>(sat, rn, s2);
  }
  {
    int tot = 32 * 128 * HO;
    k_rowsum<<<(tot + 255) / 256, 256, 0, stream>>>(s2, rowsum);
  }
  {
    int tot = 32 * HO * HO;
    k_denom<<<(tot + 255) / 256, 256, 0, stream>>>(rowsum, unc, denom);
  }

  k_corr<<<dim3(NT * NT, 4, 32), 32, 0, stream>>>(grd_pad, sat_bf, denom, pvals, pidx);
  k_final<<<1, 128, 0, stream>>>(pvals, pidx, out);
}
